// AMBAModel_55894704390394
// MI455X (gfx1250) — compile-verified, matrix operand order checked by reading, not run on hardware
//
#include <hip/hip_runtime.h>
#include <hip/hip_bf16.h>
#include <math.h>

// ---------------- model constants ----------------
#define BV     4
#define TV     1024
#define FV     128
#define DV     384
#define DEPTHV 8
#define DIV    768          // 2*DV
#define NV     16
#define DTRV   24
#define KCV    4
#define LV     513          // F/PF * T/PT + 1
#define HEADV  256
#define MTOK   (BV * LV)    // 2052 token rows
#define MPAT   (BV * (TV / 2)) // 2048 patches
#define KPAT   (FV * 2)     // 256 patch elements

typedef __bf16 bf16_t;
typedef bf16_t v16bf __attribute__((ext_vector_type(16)));
typedef bf16_t v8bf  __attribute__((ext_vector_type(8)));
typedef float  v8f   __attribute__((ext_vector_type(8)));
typedef float  v4f   __attribute__((ext_vector_type(4)));

__device__ __forceinline__ bf16_t f2bf(float f) {
    union { float f; unsigned u; } in; in.f = f;
    unsigned u = in.u;
    u += 0x7fffu + ((u >> 16) & 1u);            // round-to-nearest-even
    union { unsigned short s; bf16_t h; } out;
    out.s = (unsigned short)(u >> 16);
    return out.h;
}

__device__ __forceinline__ int imin(int a, int b) { return a < b ? a : b; }

// Stage one 8-wide k-chunk of a row into LDS as packed bf16 (b128 store).
// Fast path: two global b128 loads; fallback: guarded scalar loads (tail).
__device__ __forceinline__ void stage_row8(
    const float* __restrict__ src, int grow, int nrow,
    int gk, int kreal, bf16_t* __restrict__ dst /*16B aligned*/)
{
    v8bf pk;
    if (grow < nrow && gk + 8 <= kreal) {
        const v4f* p = (const v4f*)src;       // src already at row*ld + gk
        v4f f0 = p[0], f1 = p[1];
        #pragma unroll
        for (int e = 0; e < 4; ++e) { pk[e] = f2bf(f0[e]); pk[e + 4] = f2bf(f1[e]); }
    } else {
        #pragma unroll
        for (int e = 0; e < 8; ++e) {
            float v = (grow < nrow && gk + e < kreal) ? src[e] : 0.0f;
            pk[e] = f2bf(v);
        }
    }
    *(v8bf*)dst = pk;
}

// ---------------------------------------------------------------
// Generic GEMM: Out[M,N] = act(A[M,K] @ W[N,K]^T + bias) (+ residual)
// bf16 WMMA (v_wmma_f32_16x16x32_bf16), f32 accumulate.
// Block = 256 threads (8 wave32 waves), tile = 64x64, K-step 32.
// ACT: 0 = none, 1 = relu, 2 = softplus
// ---------------------------------------------------------------
template<int ACT, bool RES>
__global__ __launch_bounds__(256) void gemm_wmma_bf16(
    const float* __restrict__ A, int lda,
    const float* __restrict__ W,          // (Nreal, Kreal) row-major
    const float* __restrict__ bias,       // nullptr -> no bias
    const float* __restrict__ Rsd,        // residual, same layout as Out
    float* __restrict__ Out,
    int Mreal, int Nreal, int Kreal)
{
    __shared__ bf16_t As[64][32];
    __shared__ bf16_t Bs[64][32];

    const int tid  = threadIdx.x;
    const int lane = tid & 31;
    const int wid  = tid >> 5;
    const int wm   = wid & 3;
    const int wn   = wid >> 2;
    const int m0   = blockIdx.y * 64;
    const int n0   = blockIdx.x * 64;

    v8f acc0 = 0.0f;
    v8f acc1 = 0.0f;

    const int lm = lane & 15;
    const int kb = (lane >> 4) * 8;       // 0 or 8 (A/B 16-bit fragment layout)

    // staging assignment: thread -> (row, 8-wide k chunk)
    const int srow = tid >> 2;            // 0..63
    const int skc  = (tid & 3) * 8;       // 0,8,16,24
    const int gmA  = m0 + srow;
    const int gnB  = n0 + srow;
    const int clampA = imin(gmA, Mreal - 1);
    const int clampB = imin(gnB, Nreal - 1);

    for (int k0 = 0; k0 < Kreal; k0 += 32) {
        __syncthreads();

        // speculative prefetch of next K chunk -> global_prefetch_b8
        if (k0 + 32 < Kreal) {
            __builtin_prefetch(&A[(size_t)clampA * lda + k0 + 32], 0, 1);
            __builtin_prefetch(&W[(size_t)clampB * Kreal + k0 + 32], 0, 1);
        }

        // stage A tile 64(m) x 32(k): 2x global b128 -> pack -> 1x ds b128
        stage_row8(&A[(size_t)clampA * lda + k0 + skc], gmA, Mreal,
                   k0 + skc, Kreal, &As[srow][skc]);
        // stage W tile 64(n) x 32(k)
        stage_row8(&W[(size_t)clampB * Kreal + k0 + skc], gnB, Nreal,
                   k0 + skc, Kreal, &Bs[srow][skc]);

        __syncthreads();

        // A fragment: lanes 0-15 -> K {0..7,16..23}; lanes 16-31 -> K {8..15,24..31}
        v8bf alo = *(const v8bf*)&As[wm * 16 + lm][kb];
        v8bf ahi = *(const v8bf*)&As[wm * 16 + lm][kb + 16];
        v16bf afrag;
        #pragma unroll
        for (int e = 0; e < 8; ++e) { afrag[e] = alo[e]; afrag[e + 8] = ahi[e]; }

        // B fragments (column-per-lane, same K packing), two 16-wide subtiles
        v8bf blo0 = *(const v8bf*)&Bs[wn * 32 + lm][kb];
        v8bf bhi0 = *(const v8bf*)&Bs[wn * 32 + lm][kb + 16];
        v8bf blo1 = *(const v8bf*)&Bs[wn * 32 + 16 + lm][kb];
        v8bf bhi1 = *(const v8bf*)&Bs[wn * 32 + 16 + lm][kb + 16];
        v16bf bfrag0, bfrag1;
        #pragma unroll
        for (int e = 0; e < 8; ++e) {
            bfrag0[e] = blo0[e]; bfrag0[e + 8] = bhi0[e];
            bfrag1[e] = blo1[e]; bfrag1[e + 8] = bhi1[e];
        }

        acc0 = __builtin_amdgcn_wmma_f32_16x16x32_bf16(
            false, afrag, false, bfrag0, (short)0, acc0, false, false);
        acc1 = __builtin_amdgcn_wmma_f32_16x16x32_bf16(
            false, afrag, false, bfrag1, (short)0, acc1, false, false);
    }

    // epilogue: C/D layout -> lane(0..15)=N, VGPR r = M row (lanes>=16: +8)
    const int cn   = lane & 15;
    const int mofs = (lane >> 4) * 8;
    #pragma unroll
    for (int r = 0; r < 8; ++r) {
        int gm = m0 + wm * 16 + mofs + r;
        if (gm >= Mreal) continue;
        #pragma unroll
        for (int sub = 0; sub < 2; ++sub) {
            int gn = n0 + wn * 32 + sub * 16 + cn;
            if (gn >= Nreal) continue;
            float v = sub ? acc1[r] : acc0[r];
            if (bias) v += bias[gn];
            if (ACT == 1) v = fmaxf(v, 0.0f);
            if (ACT == 2) v = (v > 20.0f) ? v : logf(1.0f + __expf(v));
            if (RES) v += Rsd[(size_t)gm * Nreal + gn];
            Out[(size_t)gm * Nreal + gn] = v;
        }
    }
}

// ---------------------------------------------------------------
// Patch repack: Apat[b*512+w][i*2+j] = x[b][2w+j][i]
// ---------------------------------------------------------------
__global__ void patch_pack_k(const float* __restrict__ x, float* __restrict__ Apat)
{
    int idx = blockIdx.x * blockDim.x + threadIdx.x;
    if (idx >= MPAT * KPAT) return;
    int k = idx % KPAT;           // i*2 + j
    int m = idx / KPAT;           // b*512 + w
    int i = k >> 1, j = k & 1;
    int b = m / (TV / 2), w = m % (TV / 2);
    Apat[idx] = x[((size_t)b * TV + (2 * w + j)) * FV + i];
}

// h[b*L + 0][d] = cls[d] + pos[0][d];  h[b*L+1+w][d] = p[b*512+w][d] + pos[1+w][d]
__global__ void assemble_h_k(const float* __restrict__ p,
                             const float* __restrict__ cls,
                             const float* __restrict__ pos,
                             float* __restrict__ h)
{
    int idx = blockIdx.x * blockDim.x + threadIdx.x;
    if (idx >= MTOK * DV) return;
    int d = idx % DV;
    int m = idx / DV;
    int b = m / LV, l = m % LV;
    float v;
    if (l == 0) v = cls[d] + pos[d];
    else        v = p[((size_t)b * (TV / 2) + (l - 1)) * DV + d] + pos[(size_t)l * DV + d];
    h[idx] = v;
}

// ---------------------------------------------------------------
// LayerNorm over last dim (Drow), one block per row
// ---------------------------------------------------------------
__global__ __launch_bounds__(128) void layernorm_k(
    const float* __restrict__ X, const float* __restrict__ w,
    const float* __restrict__ b, float* __restrict__ Y, int Drow)
{
    int row = blockIdx.x;
    const float* x = X + (size_t)row * Drow;
    __shared__ float s1[128], s2[128];
    float lsum = 0.0f, lsq = 0.0f;
    for (int j = threadIdx.x; j < Drow; j += 128) {
        float v = x[j]; lsum += v; lsq += v * v;
    }
    s1[threadIdx.x] = lsum; s2[threadIdx.x] = lsq;
    __syncthreads();
    for (int st = 64; st > 0; st >>= 1) {
        if (threadIdx.x < st) {
            s1[threadIdx.x] += s1[threadIdx.x + st];
            s2[threadIdx.x] += s2[threadIdx.x + st];
        }
        __syncthreads();
    }
    float mean = s1[0] / (float)Drow;
    float var  = s2[0] / (float)Drow - mean * mean;
    float rstd = rsqrtf(var + 1e-5f);
    for (int j = threadIdx.x; j < Drow; j += 128)
        Y[(size_t)row * Drow + j] = (x[j] - mean) * rstd * w[j] + b[j];
}

// ---------------------------------------------------------------
// Depthwise causal conv1d (K=4) + SiLU. xin = xz[:, :DIV] (ld 2*DIV)
// ---------------------------------------------------------------
__global__ void conv1d_silu_k(const float* __restrict__ xz,
                              const float* __restrict__ cw,  // (DIV,4)
                              const float* __restrict__ cb,  // (DIV,)
                              float* __restrict__ xc)
{
    int idx = blockIdx.x * blockDim.x + threadIdx.x;
    if (idx >= MTOK * DIV) return;
    int c = idx % DIV;
    int m = idx / DIV;
    int b = m / LV, t = m % LV;
    float acc = cb[c];
    #pragma unroll
    for (int kk = 0; kk < KCV; ++kk) {
        int tt = t - (KCV - 1) + kk;
        if (tt >= 0)
            acc += cw[c * KCV + kk] * xz[((size_t)(b * LV + tt)) * (2 * DIV) + c];
    }
    xc[idx] = acc / (1.0f + __expf(-acc));   // silu
}

// ---------------------------------------------------------------
// Selective scan: one block per batch, one thread per channel c.
// 16-wide state in registers; B/C vectors broadcast through LDS.
// Fused: D-skip, SiLU(z) gate.
// ---------------------------------------------------------------
__global__ __launch_bounds__(DIV) void scan_k(
    const float* __restrict__ xdbl,   // (MTOK, 56): [.,24:40]=B, [.,40:56]=C
    const float* __restrict__ dt,     // (MTOK, DIV)
    const float* __restrict__ xc,     // (MTOK, DIV)
    const float* __restrict__ xz,     // (MTOK, 2*DIV): z = cols DIV..
    const float* __restrict__ Alog,   // (DIV, NV) for this layer
    const float* __restrict__ Dsk,    // (DIV,)
    float* __restrict__ Y)            // (MTOK, DIV)
{
    int b = blockIdx.x;
    int c = threadIdx.x;
    __shared__ float sB[NV], sC[NV];

    float a[NV], s[NV];
    #pragma unroll
    for (int n = 0; n < NV; ++n) {
        a[n] = -__expf(Alog[(size_t)c * NV + n]);
        s[n] = 0.0f;
    }
    float dsk = Dsk[c];

    for (int t = 0; t < LV; ++t) {
        int row = b * LV + t;
        if (c < NV)               sB[c]      = xdbl[(size_t)row * 56 + DTRV + c];
        else if (c < 2 * NV)      sC[c - NV] = xdbl[(size_t)row * 56 + DTRV + NV + (c - NV)];
        __syncthreads();

        float dtv = dt[(size_t)row * DIV + c];
        float xcv = xc[(size_t)row * DIV + c];
        float yacc = 0.0f;
        #pragma unroll
        for (int n = 0; n < NV; ++n) {
            s[n] = __expf(dtv * a[n]) * s[n] + dtv * sB[n] * xcv;
            yacc += s[n] * sC[n];
        }
        float zv = xz[(size_t)row * (2 * DIV) + DIV + c];
        Y[(size_t)row * DIV + c] = (yacc + dsk * xcv) * (zv / (1.0f + __expf(-zv)));
        __syncthreads();
    }
}

// ---------------------------------------------------------------
extern "C" void kernel_launch(void* const* d_in, const int* in_sizes, int n_in,
                              void* d_out, int out_size, void* d_ws, size_t ws_size,
                              hipStream_t stream)
{
    const float* x         = (const float*)d_in[0];
    const float* conv_w    = (const float*)d_in[1];
    const float* conv_b    = (const float*)d_in[2];
    const float* cls_tok   = (const float*)d_in[3];
    const float* pos_emb   = (const float*)d_in[4];
    const float* norm_w    = (const float*)d_in[5];
    const float* norm_b    = (const float*)d_in[6];
    const float* in_proj_w = (const float*)d_in[7];
    const float* conv1d_w  = (const float*)d_in[8];
    const float* conv1d_b  = (const float*)d_in[9];
    const float* x_proj_w  = (const float*)d_in[10];
    const float* dt_proj_w = (const float*)d_in[11];
    const float* dt_proj_b = (const float*)d_in[12];
    const float* A_log     = (const float*)d_in[13];
    const float* D_skip    = (const float*)d_in[14];
    const float* out_proj_w= (const float*)d_in[15];
    const float* normf_w   = (const float*)d_in[16];
    const float* normf_b   = (const float*)d_in[17];
    const float* head_w1   = (const float*)d_in[18];
    const float* head_b1   = (const float*)d_in[19];
    const float* head_w2   = (const float*)d_in[20];
    const float* head_b2   = (const float*)d_in[21];
    float* out = (float*)d_out;

    float* ws = (float*)d_ws;
    size_t off = 0;
    auto carve = [&](size_t nflt) {
        float* p = ws + off;
        off += (nflt + 63) & ~(size_t)63;
        return p;
    };
    float* xzb  = carve((size_t)MTOK * 2 * DIV);  // (2052,1536); also hosts Apat+p pre-loop
    float* h    = carve((size_t)MTOK * DV);
    float* hn   = carve((size_t)MTOK * DV);
    float* xc   = carve((size_t)MTOK * DIV);
    float* xdbl = carve((size_t)MTOK * 56);
    float* dtb  = carve((size_t)MTOK * DIV);
    float* yb   = carve((size_t)MTOK * DIV);
    float* Apat = xzb;                             // 2048*256 = 524288 floats
    float* pbuf = xzb + (size_t)MPAT * KPAT;       // 2048*384, fits inside xz region
    float* h1   = yb;                              // head hidden reuses y (dead by then)

    dim3 blk(256);

    // --- patch embed as GEMM ---
    patch_pack_k<<<dim3((MPAT * KPAT + 255) / 256), blk, 0, stream>>>(x, Apat);
    {
        dim3 g((DV + 63) / 64, (MPAT + 63) / 64);
        gemm_wmma_bf16<0, false><<<g, blk, 0, stream>>>(
            Apat, KPAT, conv_w, conv_b, nullptr, pbuf, MPAT, DV, KPAT);
    }
    assemble_h_k<<<dim3((MTOK * DV + 255) / 256), blk, 0, stream>>>(pbuf, cls_tok, pos_emb, h);

    // --- Mamba blocks ---
    const int gM = (MTOK + 63) / 64;   // 33
    for (int i = 0; i < DEPTHV; ++i) {
        layernorm_k<<<dim3(MTOK), dim3(128), 0, stream>>>(
            h, norm_w + i * DV, norm_b + i * DV, hn, DV);

        gemm_wmma_bf16<0, false><<<dim3((2 * DIV + 63) / 64, gM), blk, 0, stream>>>(
            hn, DV, in_proj_w + (size_t)i * 2 * DIV * DV, nullptr, nullptr,
            xzb, MTOK, 2 * DIV, DV);

        conv1d_silu_k<<<dim3((MTOK * DIV + 255) / 256), blk, 0, stream>>>(
            xzb, conv1d_w + (size_t)i * DIV * KCV, conv1d_b + (size_t)i * DIV, xc);

        gemm_wmma_bf16<0, false><<<dim3((56 + 63) / 64, gM), blk, 0, stream>>>(
            xc, DIV, x_proj_w + (size_t)i * 56 * DIV, nullptr, nullptr,
            xdbl, MTOK, 56, DIV);

        gemm_wmma_bf16<2, false><<<dim3((DIV + 63) / 64, gM), blk, 0, stream>>>(
            xdbl, 56, dt_proj_w + (size_t)i * DIV * DTRV, dt_proj_b + (size_t)i * DIV,
            nullptr, dtb, MTOK, DIV, DTRV);

        scan_k<<<dim3(BV), dim3(DIV), 0, stream>>>(
            xdbl, dtb, xc, xzb, A_log + (size_t)i * DIV * NV, D_skip + (size_t)i * DIV, yb);

        gemm_wmma_bf16<0, true><<<dim3((DV + 63) / 64, gM), blk, 0, stream>>>(
            yb, DIV, out_proj_w + (size_t)i * DV * DIV, nullptr, h,
            h, MTOK, DV, DIV);
    }

    // --- final norm + head ---
    layernorm_k<<<dim3(MTOK), dim3(128), 0, stream>>>(h, normf_w, normf_b, hn, DV);

    gemm_wmma_bf16<1, false><<<dim3((DV + 63) / 64, gM), blk, 0, stream>>>(
        hn, DV, head_w1, head_b1, nullptr, h1, MTOK, DV, DV);

    gemm_wmma_bf16<0, false><<<dim3((HEADV + 63) / 64, gM), blk, 0, stream>>>(
        h1, DV, head_w2, head_b2, nullptr, out, MTOK, HEADV, DV);
}
